// AttentionTSSA_28372553957437
// MI455X (gfx1250) — matrix operand order repaired and checked by reading, not compiled, not run on hardware
//
#include <hip/hip_runtime.h>
#include <hip/hip_bf16.h>

// ---------------------------------------------------------------------------
// AttentionTSSA for MI455X (gfx1250, wave32, WMMA).
//
// Pipeline (all in the [B, C, N] "transposed" layout so no tensor transposes
// are ever materialized):
//   1) wT  = W_qkv @ x_b          (bf16x3 WMMA GEMM, per batch 768x768 * 768x4096)
//   2) invn2[b,c]   = 1/max(sum_n wT^2, 1e-24)          (token-axis F.normalize)
//   3) Pi[b,h,n]    = softmax_h( temp[h] * sum_d wT^2 * invn2 )
//   4) S[b,h]       = sum_n Pi
//   5) attn[b,c]    = 1/(1 + (sum_n Pi * wT^2)/(S+1e-8))
//   6) wT          <- -wT * Pi * attn                    (in-place, becomes y)
//   7) out = W_out @ y_b          (same WMMA GEMM, writes [B,C,H,W] directly)
// ---------------------------------------------------------------------------

typedef __attribute__((ext_vector_type(16))) __bf16 v16bf;
typedef __attribute__((ext_vector_type(8)))  __bf16 v8bf;
typedef __attribute__((ext_vector_type(8)))  float  v8f;

#define BATCH 16
#define CCH   768
#define NTOK  4096
#define HEADS 12
#define DHEAD 64

#define BM 128      // block tile rows (c)
#define BN 128      // block tile cols (n)
#define KC 32       // fp32 K per chunk
#define LDK 72      // LDS row stride in bf16 (64 data = hi|lo, +8 pad -> 16B aligned frags)

union FragB16 { v16bf v; v8bf h[2]; };

// D[c,n] = sum_k Wm[c,k] * X[b,k,n], fp32 in/out, bf16x3 WMMA core.
__global__ __launch_bounds__(256)
void gemm_bf16x3_kernel(const float* __restrict__ Wm,   // [768][768], k contiguous
                        const float* __restrict__ X,    // [B][768][4096], n contiguous
                        float* __restrict__ Out)        // [B][768][4096]
{
    __shared__ __attribute__((aligned(16))) __bf16 ldsA[BM * LDK];
    __shared__ __attribute__((aligned(16))) __bf16 ldsB[BN * LDK];   // transposed: [n][k]

    const int tid  = threadIdx.x;
    const int lane = tid & 31;
    const int wave = tid >> 5;       // 0..7
    const int wm   = wave & 1;       // M half (64 rows)
    const int wn   = wave >> 1;      // N quarter (32 cols)
    const int nb   = blockIdx.x * BN;
    const int cb   = blockIdx.y * BM;
    const int bz   = blockIdx.z;

    const float* Xb = X   + (size_t)bz * CCH * NTOK;
    float*       Ob = Out + (size_t)bz * CCH * NTOK;

    const int lm   = lane & 15;
    const int hilo = lane >> 4;      // 0 / 1
    const int kb8  = hilo * 8;       // unused (A frag also uses kb16)
    const int kb16 = hilo * 16;      // B frag: K 0-15 / 16-31

    v8f acc[4][2] = {};

    for (int k0 = 0; k0 < CCH; k0 += KC) {
        // ---- fill A tile: 128 x 32 fp32 -> hi|lo bf16 segments ----
        #pragma unroll
        for (int i = 0; i < 16; ++i) {
            int idx = tid + i * 256;           // 0..4095
            int m = idx >> 5;                  // 0..127
            int k = idx & 31;                  // 0..31
            float v = Wm[(size_t)(cb + m) * CCH + k0 + k];
            __bf16 hi = (__bf16)v;
            __bf16 lo = (__bf16)(v - (float)hi);
            ldsA[m * LDK + k]      = hi;
            ldsA[m * LDK + 32 + k] = lo;
        }
        // ---- fill B tile transposed: global [k][n] -> lds [n][hi|lo k] ----
        #pragma unroll
        for (int i = 0; i < 16; ++i) {
            int idx = tid + i * 256;
            int k = idx >> 7;                  // 0..31
            int n = idx & 127;                 // 0..127
            float v = Xb[(size_t)(k0 + k) * NTOK + nb + n];
            __bf16 hi = (__bf16)v;
            __bf16 lo = (__bf16)(v - (float)hi);
            ldsB[n * LDK + k]      = hi;
            ldsB[n * LDK + 32 + k] = lo;
        }
        // prefetch next K chunk of X into cache (global_prefetch_b8)
        if (k0 + KC < CCH) {
            __builtin_prefetch(&Xb[(size_t)(k0 + KC + (tid >> 3)) * NTOK + nb + ((tid & 7) << 4)], 0, 0);
        }
        __syncthreads();

        // ---- bf16x3: (hi+lo)*(hi+lo) ~= hi*hi + lo*hi + hi*lo ----
        #pragma unroll
        for (int s = 0; s < 3; ++s) {
            const int segA = (s == 1) ? 32 : 0;
            const int segB = (s == 2) ? 32 : 0;
            FragB16 a[4], b[2];
            #pragma unroll
            for (int t = 0; t < 4; ++t) {
                const __bf16* p = &ldsA[(wm * 64 + t * 16 + lm) * LDK + segA + kb16];
                a[t].h[0] = *(const v8bf*)(p);        // K kb16..kb16+7
                a[t].h[1] = *(const v8bf*)(p + 8);    // K kb16+8..kb16+15
            }
            #pragma unroll
            for (int u = 0; u < 2; ++u) {
                const __bf16* p = &ldsB[(wn * 32 + u * 16 + lm) * LDK + segB + kb16];
                b[u].h[0] = *(const v8bf*)(p);        // K kb16..kb16+7
                b[u].h[1] = *(const v8bf*)(p + 8);    // K kb16+8..kb16+15
            }
            #pragma unroll
            for (int t = 0; t < 4; ++t)
                #pragma unroll
                for (int u = 0; u < 2; ++u)
                    acc[t][u] = __builtin_amdgcn_wmma_f32_16x16x32_bf16(
                        false, a[t].v, false, b[u].v, (short)0, acc[t][u], false, false);
        }
        __syncthreads();
    }

    // ---- epilogue: C/D layout -> lane holds col lm, rows hilo*8 + r ----
    #pragma unroll
    for (int t = 0; t < 4; ++t) {
        #pragma unroll
        for (int u = 0; u < 2; ++u) {
            const int col  = nb + wn * 32 + u * 16 + lm;
            const int mrow = cb + wm * 64 + t * 16 + hilo * 8;
            #pragma unroll
            for (int r = 0; r < 8; ++r)
                Ob[(size_t)(mrow + r) * NTOK + col] = acc[t][u][r];
        }
    }
}

// ---------------------------------------------------------------------------
__device__ __forceinline__ float blockReduceSum(float v, float* red, int tid) {
    red[tid] = v;
    __syncthreads();
    #pragma unroll
    for (int s = 128; s >= 1; s >>= 1) {
        if (tid < s) red[tid] += red[tid + s];
        __syncthreads();
    }
    return red[0];
}

// invn2[b,c] = 1 / max(sum_n wT[b,c,n]^2, 1e-24)
__global__ __launch_bounds__(256)
void colnorm_kernel(const float* __restrict__ wT, float* __restrict__ invn2) {
    __shared__ float red[256];
    const int c = blockIdx.x, b = blockIdx.y, tid = threadIdx.x;
    const float* p = wT + ((size_t)b * CCH + c) * NTOK;
    float s = 0.f;
    for (int n = tid; n < NTOK; n += 256) { float v = p[n]; s += v * v; }
    s = blockReduceSum(s, red, tid);
    if (tid == 0) invn2[b * CCH + c] = 1.0f / fmaxf(s, 1e-24f);
}

// Pi[b,h,n] = softmax_h( temp[h] * sum_d wT^2 * invn2 )
__global__ __launch_bounds__(256)
void softmax_heads_kernel(const float* __restrict__ wT, const float* __restrict__ invn2,
                          const float* __restrict__ temp, float* __restrict__ Pi) {
    __shared__ float sinv[CCH];
    __shared__ float stmp[HEADS];
    const int b = blockIdx.y;
    const int n = blockIdx.x * 256 + threadIdx.x;
    for (int i = threadIdx.x; i < CCH; i += 256) sinv[i] = invn2[b * CCH + i];
    if (threadIdx.x < HEADS) stmp[threadIdx.x] = temp[threadIdx.x];
    __syncthreads();

    float l[HEADS];
    float m = -1e30f;
    #pragma unroll
    for (int h = 0; h < HEADS; ++h) {
        float acc = 0.f;
        #pragma unroll 8
        for (int dd = 0; dd < DHEAD; ++dd) {
            const int c = h * DHEAD + dd;
            float v = wT[((size_t)b * CCH + c) * NTOK + n];
            acc += v * v * sinv[c];
        }
        l[h] = acc * stmp[h];
        m = fmaxf(m, l[h]);
    }
    float sum = 0.f;
    #pragma unroll
    for (int h = 0; h < HEADS; ++h) { l[h] = __expf(l[h] - m); sum += l[h]; }
    const float inv = 1.0f / sum;
    #pragma unroll
    for (int h = 0; h < HEADS; ++h)
        Pi[((size_t)b * HEADS + h) * NTOK + n] = l[h] * inv;
}

// S[b,h] = sum_n Pi[b,h,n]
__global__ __launch_bounds__(256)
void head_sum_kernel(const float* __restrict__ Pi, float* __restrict__ S) {
    __shared__ float red[256];
    const int h = blockIdx.x, b = blockIdx.y, tid = threadIdx.x;
    const float* p = Pi + ((size_t)b * HEADS + h) * NTOK;
    float s = 0.f;
    for (int n = tid; n < NTOK; n += 256) s += p[n];
    s = blockReduceSum(s, red, tid);
    if (tid == 0) S[b * HEADS + h] = s;
}

// attn[b,c] = 1 / (1 + (sum_n Pi * wT^2) / (S + 1e-8))
__global__ __launch_bounds__(256)
void dots_attn_kernel(const float* __restrict__ wT, const float* __restrict__ Pi,
                      const float* __restrict__ S, float* __restrict__ attn) {
    __shared__ float red[256];
    const int c = blockIdx.x, b = blockIdx.y, tid = threadIdx.x, h = c >> 6;
    const float* pw = wT + ((size_t)b * CCH + c) * NTOK;
    const float* pp = Pi + ((size_t)b * HEADS + h) * NTOK;
    float s = 0.f;
    for (int n = tid; n < NTOK; n += 256) { float v = pw[n]; s += pp[n] * v * v; }
    s = blockReduceSum(s, red, tid);
    if (tid == 0) {
        const float dots = s / (S[b * HEADS + h] + 1e-8f);
        attn[b * CCH + c] = 1.0f / (1.0f + dots);
    }
}

// wT <- -wT * Pi * attn   (in place; vectorized float4)
__global__ __launch_bounds__(256)
void scale_kernel(float* __restrict__ wT, const float* __restrict__ Pi,
                  const float* __restrict__ attn) {
    const size_t i = (size_t)blockIdx.x * 256 + threadIdx.x;   // over B*C*(N/4)
    const int nq = (int)(i & 1023);                            // N/4 = 1024
    const size_t t = i >> 10;
    const int c = (int)(t % CCH);
    const int b = (int)(t / CCH);
    const int h = c >> 6;
    const float a = attn[b * CCH + c];
    float4* pw = (float4*)(wT + ((size_t)b * CCH + c) * NTOK);
    const float4* pp = (const float4*)(Pi + ((size_t)b * HEADS + h) * NTOK);
    float4 w = pw[nq];
    float4 p = pp[nq];
    w.x = -w.x * p.x * a;
    w.y = -w.y * p.y * a;
    w.z = -w.z * p.z * a;
    w.w = -w.w * p.w * a;
    pw[nq] = w;
}

// ---------------------------------------------------------------------------
extern "C" void kernel_launch(void* const* d_in, const int* in_sizes, int n_in,
                              void* d_out, int out_size, void* d_ws, size_t ws_size,
                              hipStream_t stream) {
    (void)in_sizes; (void)n_in; (void)out_size; (void)ws_size;
    const float* x    = (const float*)d_in[0];   // [16,768,64,64] == [B][C][N]
    const float* Wqkv = (const float*)d_in[1];   // [768,768]
    const float* Wout = (const float*)d_in[2];   // [768,768]
    const float* temp = (const float*)d_in[3];   // [12,1]
    float* out = (float*)d_out;                  // [16,768,64,64]

    // workspace carve-up (all fully rewritten every call; ~205 MB)
    float* ws    = (float*)d_ws;
    const size_t NW = (size_t)BATCH * CCH * NTOK;          // 50,331,648
    float* wT    = ws;                                     // [B][C][N]
    float* Pi    = wT + NW;                                // [B][12][N]
    float* invn2 = Pi + (size_t)BATCH * HEADS * NTOK;      // [B][C]
    float* S     = invn2 + BATCH * CCH;                    // [B][12]
    float* attn  = S + BATCH * HEADS;                      // [B][C]

    const dim3 gGemm(NTOK / BN, CCH / BM, BATCH);          // (32, 6, 16)

    gemm_bf16x3_kernel<<<gGemm, 256, 0, stream>>>(Wqkv, x, wT);
    colnorm_kernel<<<dim3(CCH, BATCH), 256, 0, stream>>>(wT, invn2);
    softmax_heads_kernel<<<dim3(NTOK / 256, BATCH), 256, 0, stream>>>(wT, invn2, temp, Pi);
    head_sum_kernel<<<dim3(HEADS, BATCH), 256, 0, stream>>>(Pi, S);
    dots_attn_kernel<<<dim3(CCH, BATCH), 256, 0, stream>>>(wT, Pi, S, attn);
    scale_kernel<<<(unsigned)((size_t)BATCH * CCH * (NTOK / 4) / 256), 256, 0, stream>>>(wT, Pi, attn);
    gemm_bf16x3_kernel<<<gGemm, 256, 0, stream>>>(Wout, wT, out);
}
